// MultiHeadAttn_51917564674332
// MI455X (gfx1250) — matrix operand order repaired and checked by reading, not compiled, not run on hardware
//
#include <hip/hip_runtime.h>
#include <stdint.h>
#include <stddef.h>

// ---------------------------------------------------------------------------
// MI455X (gfx1250) implementation of TXL multi-head attention block.
// wave32, WMMA bf16 (f32 accumulate) for all matrix math, flash-softmax.
// fp32 operands are pre-converted to bf16 ONCE; GEMM inner loops stream pure
// bf16 through double-buffered LDS tiles into v_wmma_f32_16x16x32_bf16.
// ---------------------------------------------------------------------------

#define T_TOK  2048
#define BSZ    4
#define NHEAD  16
#define DHEAD  64
#define DMODEL 1024
#define NROWS  (T_TOK * BSZ)        // 8192 rows of (T,B,*) flattened
#define NQKV   (3 * DMODEL)         // 3072 fused qkv output features
#define ATT_SCALE 0.125f            // 1/sqrt(64)

typedef __attribute__((ext_vector_type(16))) __bf16    bf16x16;
typedef __attribute__((ext_vector_type(2)))  __bf16    bf16x2;
typedef __attribute__((ext_vector_type(8)))  float     f32x8;
typedef __attribute__((ext_vector_type(4)))  float     f32x4;
typedef __attribute__((ext_vector_type(4)))  uint32_t  u32x4;
typedef __attribute__((ext_vector_type(2)))  uint32_t  u32x2;

union FragBF { u32x4 q[2]; bf16x16 v; };   // 32B: one WMMA A/B operand

__device__ __forceinline__ uint32_t pack2bf(float a, float b) {
#if __has_builtin(__builtin_amdgcn_cvt_pk_bf16_f32)
  union { bf16x2 v; uint32_t u; } r;
  r.v = __builtin_amdgcn_cvt_pk_bf16_f32(a, b);
  return r.u;
#else
  union { float f; uint32_t u; } x, y; x.f = a; y.f = b;
  uint32_t ua = x.u + (0x7fffu + ((x.u >> 16) & 1u));   // RNE
  uint32_t ub = y.u + (0x7fffu + ((y.u >> 16) & 1u));
  return (ua >> 16) | (ub & 0xffff0000u);
#endif
}
__device__ __forceinline__ unsigned short f2bf(float f) {
  return (unsigned short)(pack2bf(f, 0.0f) & 0xffffu);
}

// ---------------------------------------------------------------------------
// Kernel 0: one-shot fp32 -> bf16 conversion (bandwidth bound, 8 elems/thread)
// ---------------------------------------------------------------------------
__global__ __launch_bounds__(256)
void cvt_bf16_kernel(const float* __restrict__ src,
                     unsigned short* __restrict__ dst, int n)
{
  const int i = (blockIdx.x * 256 + threadIdx.x) * 8;
  if (i + 8 <= n) {
    const f32x4 a = *(const f32x4*)(src + i);
    const f32x4 b = *(const f32x4*)(src + i + 4);
    u32x4 o;
    o.x = pack2bf(a.x, a.y); o.y = pack2bf(a.z, a.w);
    o.z = pack2bf(b.x, b.y); o.w = pack2bf(b.z, b.w);
    *(u32x4*)(dst + i) = o;
  }
}

// ---------------------------------------------------------------------------
// Kernel 1: fused QKV projection.  C[m,n] = sum_k hb[m,k] * Wqkv[n,k]  (bf16)
// Block tile 128(M) x 64(N), 8 waves each own 16x64, K step 32, LDS ping-pong.
// Stores Q,K as [b,h,t,d] and V transposed as [b,h,d,t] (bf16) so every WMMA
// operand in the attention kernel is a contiguous per-lane 16B load.
// ---------------------------------------------------------------------------
__global__ __launch_bounds__(256)
void qkv_proj_kernel(const unsigned short* __restrict__ hb,
                     const unsigned short* __restrict__ wqkvb,
                     unsigned short* __restrict__ Qws,
                     unsigned short* __restrict__ Kws,
                     unsigned short* __restrict__ VTws)
{
  __shared__ unsigned short Abuf[2][128 * 32];   // [m][k] bf16, 8KB x2
  __shared__ unsigned short Bbuf[2][64 * 32];    // [n][k] bf16, 4KB x2
  const int tid  = threadIdx.x;
  const int lane = tid & 31, w = tid >> 5;
  const int lm   = lane & 15, half = lane >> 4;
  const int m0   = blockIdx.x * 128;
  const int n0   = blockIdx.y * 64;

  // cooperative-tile indices (constant across the k loop)
  const int eA = tid * 16, mA = eA >> 5, kA = eA & 31;   // A: 16 elems/thread
  const int eB = tid * 8,  nB = eB >> 5, kB = eB & 31;   // B:  8 elems/thread
  const unsigned short* aSrc = hb    + (size_t)(m0 + mA) * DMODEL + kA;
  const unsigned short* bSrc = wqkvb + (size_t)(n0 + nB) * DMODEL + kB;

  f32x8 acc[4] = {};

  // prologue: tile 0 into regs, then LDS buffer 0
  u32x4 ar0 = *(const u32x4*)(aSrc);
  u32x4 ar1 = *(const u32x4*)(aSrc + 8);
  u32x4 br0 = *(const u32x4*)(bSrc);
  *(u32x4*)&Abuf[0][mA * 32 + kA]     = ar0;
  *(u32x4*)&Abuf[0][mA * 32 + kA + 8] = ar1;
  *(u32x4*)&Bbuf[0][nB * 32 + kB]     = br0;

  int cur = 0;
  for (int k0 = 0; k0 < DMODEL; k0 += 32) {
    __syncthreads();                      // buf[cur] ready for all waves
    const bool more = (k0 + 32) < DMODEL;
    if (more) {                           // issue next-tile loads early
      ar0 = *(const u32x4*)(aSrc + k0 + 32);
      ar1 = *(const u32x4*)(aSrc + k0 + 40);
      br0 = *(const u32x4*)(bSrc + k0 + 32);
    }

    FragBF a;                             // A frag: row m, 16 contiguous k
    const int mrow = w * 16 + lm;
    a.q[0] = *(const u32x4*)&Abuf[cur][mrow * 32 + half * 16];
    a.q[1] = *(const u32x4*)&Abuf[cur][mrow * 32 + half * 16 + 8];
#pragma unroll
    for (int t = 0; t < 4; ++t) {         // B frag: col n, 16 contiguous k
      FragBF bf;
      const int n = t * 16 + lm;
      bf.q[0] = *(const u32x4*)&Bbuf[cur][n * 32 + half * 16];
      bf.q[1] = *(const u32x4*)&Bbuf[cur][n * 32 + half * 16 + 8];
      acc[t] = __builtin_amdgcn_wmma_f32_16x16x32_bf16(
          false, a.v, false, bf.v, (short)0, acc[t], false, false);
    }

    if (more) {                           // stage next tile into other buffer
      *(u32x4*)&Abuf[cur ^ 1][mA * 32 + kA]     = ar0;
      *(u32x4*)&Abuf[cur ^ 1][mA * 32 + kA + 8] = ar1;
      *(u32x4*)&Bbuf[cur ^ 1][nB * 32 + kB]     = br0;
    }
    cur ^= 1;
  }

  // Epilogue: region decision is UNIFORM per block (64 | 1024).
  const int region = n0 >> 10;          // 0 = Q, 1 = K, 2 = V
  const int nbase  = n0 & (DMODEL - 1);
  if (region == 0) {
#pragma unroll
    for (int t = 0; t < 4; ++t)
#pragma unroll
      for (int r = 0; r < 8; ++r) {
        const int mg = m0 + w * 16 + r + 8 * half;
        const int nf = nbase + t * 16 + lm;
        const int tok = mg >> 2, b = mg & 3, hh = nf >> 6, d = nf & 63;
        Qws[(((size_t)(b * NHEAD + hh)) * T_TOK + tok) * DHEAD + d] = f2bf(acc[t][r]);
      }
  } else if (region == 1) {
#pragma unroll
    for (int t = 0; t < 4; ++t)
#pragma unroll
      for (int r = 0; r < 8; ++r) {
        const int mg = m0 + w * 16 + r + 8 * half;
        const int nf = nbase + t * 16 + lm;
        const int tok = mg >> 2, b = mg & 3, hh = nf >> 6, d = nf & 63;
        Kws[(((size_t)(b * NHEAD + hh)) * T_TOK + tok) * DHEAD + d] = f2bf(acc[t][r]);
      }
  } else {
#pragma unroll
    for (int t = 0; t < 4; ++t)
#pragma unroll
      for (int r = 0; r < 8; ++r) {
        const int mg = m0 + w * 16 + r + 8 * half;
        const int nf = nbase + t * 16 + lm;
        const int tok = mg >> 2, b = mg & 3, hh = nf >> 6, d = nf & 63;
        VTws[(((size_t)(b * NHEAD + hh)) * DHEAD + d) * T_TOK + tok] = f2bf(acc[t][r]);
      }
  }
}

// ---------------------------------------------------------------------------
// Kernel 2: flash attention.  Block = 4 waves sharing one (b,h); wave w owns
// 16 query rows.  K/V tiles staged once per block in double-buffered LDS;
// per 32-key step: S = Q K^T (4 wmma), fp32 online softmax, P through
// per-wave LDS (C-layout -> A-operand layout), O += P V (4 wmma).
// ---------------------------------------------------------------------------
__global__ __launch_bounds__(128)
void flash_attn_kernel(const unsigned short* __restrict__ Qws,
                       const unsigned short* __restrict__ Kws,
                       const unsigned short* __restrict__ VTws,
                       const unsigned char* __restrict__ mask,   // (T,B) bool
                       unsigned short* __restrict__ AVws)        // (T,B,1024) bf16
{
  __shared__ unsigned short Kbuf[2][32 * 64];   // [j][d], 4KB x2
  __shared__ unsigned short Vbuf[2][64 * 32];   // [d][j], 4KB x2
  __shared__ unsigned short Pbuf[4][16 * 32];   // per-wave P tile, 1KB each
  const int tid  = threadIdx.x;
  const int lane = tid & 31, w = tid >> 5;
  const int lm   = lane & 15, half = lane >> 4;
  const int bh   = blockIdx.y;                  // b*NHEAD + h
  const int b    = bh >> 4, hh = bh & 15;
  const int i0   = blockIdx.x * 64 + w * 16;

  const unsigned short* Kbase = Kws  + ((size_t)bh) * T_TOK * DHEAD;
  const unsigned short* Vbase = VTws + ((size_t)bh) * DHEAD * T_TOK;

  // Q fragments for row m = i0+lm, d = 0..31 and 32..63 (loaded once)
  FragBF aq0, aq1;
  {
    const unsigned short* qrow =
        Qws + (((size_t)bh) * T_TOK + (i0 + lm)) * DHEAD;
    aq0.q[0] = *(const u32x4*)(qrow + half * 16);
    aq0.q[1] = *(const u32x4*)(qrow + half * 16 + 8);
    aq1.q[0] = *(const u32x4*)(qrow + 32 + half * 16);
    aq1.q[1] = *(const u32x4*)(qrow + 32 + half * 16 + 8);
  }

  f32x8 o[4] = {};
  float mrun[8], lrun[8];
#pragma unroll
  for (int r = 0; r < 8; ++r) { mrun[r] = -__builtin_inff(); lrun[r] = 0.0f; }

  // cooperative K/V tile indices (constant across loop)
  const int ke = tid * 16;
  const int kj = ke >> 6, kd = ke & 63;        // K tile: 32x64
  const int vd = ke >> 5, vj = ke & 31;        // V tile: 64x32
  const unsigned short* kSrc = Kbase + (size_t)kj * DHEAD + kd;  // + j0*DHEAD
  const unsigned short* vSrc = Vbase + (size_t)vd * T_TOK + vj;  // + j0

  // prologue: tile 0 into regs, then LDS buffer 0
  u32x4 kr0 = *(const u32x4*)(kSrc);
  u32x4 kr1 = *(const u32x4*)(kSrc + 8);
  u32x4 vr0 = *(const u32x4*)(vSrc);
  u32x4 vr1 = *(const u32x4*)(vSrc + 8);
  *(u32x4*)&Kbuf[0][kj * 64 + kd]     = kr0;
  *(u32x4*)&Kbuf[0][kj * 64 + kd + 8] = kr1;
  *(u32x4*)&Vbuf[0][vd * 32 + vj]     = vr0;
  *(u32x4*)&Vbuf[0][vd * 32 + vj + 8] = vr1;

  int cur = 0;
  for (int j0 = 0; j0 < T_TOK; j0 += 32) {
    __syncthreads();                    // buf[cur] visible to all waves
    const bool more = (j0 + 32) < T_TOK;
    if (more) {                         // issue next-tile global loads early
      kr0 = *(const u32x4*)(kSrc + (size_t)(j0 + 32) * DHEAD);
      kr1 = *(const u32x4*)(kSrc + (size_t)(j0 + 32) * DHEAD + 8);
      vr0 = *(const u32x4*)(vSrc + j0 + 32);
      vr1 = *(const u32x4*)(vSrc + j0 + 40);
    }

    // ---- scores: two 16x16 tiles (j0..j0+15, j0+16..j0+31), K-dim = d = 64
    f32x8 s0 = {}, s1 = {};
    {
      FragBF k0, k1;                                  // rows j0+lm
      k0.q[0] = *(const u32x4*)&Kbuf[cur][lm * 64 + half * 16];
      k0.q[1] = *(const u32x4*)&Kbuf[cur][lm * 64 + half * 16 + 8];
      k1.q[0] = *(const u32x4*)&Kbuf[cur][lm * 64 + 32 + half * 16];
      k1.q[1] = *(const u32x4*)&Kbuf[cur][lm * 64 + 32 + half * 16 + 8];
      s0 = __builtin_amdgcn_wmma_f32_16x16x32_bf16(false, aq0.v, false, k0.v,
                                                   (short)0, s0, false, false);
      s0 = __builtin_amdgcn_wmma_f32_16x16x32_bf16(false, aq1.v, false, k1.v,
                                                   (short)0, s0, false, false);
    }
    {
      FragBF k0, k1;                                  // rows j0+16+lm
      const int rb = (16 + lm) * 64;
      k0.q[0] = *(const u32x4*)&Kbuf[cur][rb + half * 16];
      k0.q[1] = *(const u32x4*)&Kbuf[cur][rb + half * 16 + 8];
      k1.q[0] = *(const u32x4*)&Kbuf[cur][rb + 32 + half * 16];
      k1.q[1] = *(const u32x4*)&Kbuf[cur][rb + 32 + half * 16 + 8];
      s1 = __builtin_amdgcn_wmma_f32_16x16x32_bf16(false, aq0.v, false, k0.v,
                                                   (short)0, s1, false, false);
      s1 = __builtin_amdgcn_wmma_f32_16x16x32_bf16(false, aq1.v, false, k1.v,
                                                   (short)0, s1, false, false);
    }

    // ---- scale + key-padding mask (lane's column j is fixed per tile)
    const bool msk0 = mask[(size_t)(j0 + lm) * BSZ + b] != 0;
    const bool msk1 = mask[(size_t)(j0 + 16 + lm) * BSZ + b] != 0;

    float mnew[8], corr[8];
#pragma unroll
    for (int r = 0; r < 8; ++r) {
      const float a0 = msk0 ? -1e30f : s0[r] * ATT_SCALE;
      const float a1 = msk1 ? -1e30f : s1[r] * ATT_SCALE;
      s0[r] = a0; s1[r] = a1;
      float v = fmaxf(a0, a1);
#pragma unroll
      for (int off = 8; off >= 1; off >>= 1)      // 16-lane row reduction
        v = fmaxf(v, __shfl_xor(v, off, 32));
      mnew[r] = fmaxf(mrun[r], v);
      corr[r] = __expf(mrun[r] - mnew[r]);
      mrun[r] = mnew[r];
    }

    // ---- P = exp(S - m), row sums, stage P into LDS (A-operand layout)
#pragma unroll
    for (int r = 0; r < 8; ++r) {
      const float p0 = __expf(s0[r] - mnew[r]);
      const float p1 = __expf(s1[r] - mnew[r]);
      float ps = p0 + p1;
#pragma unroll
      for (int off = 8; off >= 1; off >>= 1)
        ps += __shfl_xor(ps, off, 32);
      lrun[r] = lrun[r] * corr[r] + ps;
      const int mloc = r + 8 * half;
      Pbuf[w][mloc * 32 + lm]      = f2bf(p0);
      Pbuf[w][mloc * 32 + 16 + lm] = f2bf(p1);
#pragma unroll
      for (int t = 0; t < 4; ++t) o[t][r] *= corr[r];
    }
    // Per-wave LDS round-trip: DS ops of one wave are processed in order,
    // but make the dependency explicit and keep the compiler from reordering.
    asm volatile("s_wait_dscnt 0" ::: "memory");

    FragBF ap;   // P as A operand: 16(i) x 32(j)
    ap.q[0] = *(const u32x4*)&Pbuf[w][lm * 32 + half * 16];
    ap.q[1] = *(const u32x4*)&Pbuf[w][lm * 32 + half * 16 + 8];

    // ---- O += P V : B operand from LDS V^T tile, contiguous in j
#pragma unroll
    for (int t = 0; t < 4; ++t) {
      FragBF bv;
      const int vb = (t * 16 + lm) * 32;
      bv.q[0] = *(const u32x4*)&Vbuf[cur][vb + half * 16];
      bv.q[1] = *(const u32x4*)&Vbuf[cur][vb + half * 16 + 8];
      o[t] = __builtin_amdgcn_wmma_f32_16x16x32_bf16(
          false, ap.v, false, bv.v, (short)0, o[t], false, false);
    }

    if (more) {                          // stage next tile into other buffer
      *(u32x4*)&Kbuf[cur ^ 1][kj * 64 + kd]     = kr0;
      *(u32x4*)&Kbuf[cur ^ 1][kj * 64 + kd + 8] = kr1;
      *(u32x4*)&Vbuf[cur ^ 1][vd * 32 + vj]     = vr0;
      *(u32x4*)&Vbuf[cur ^ 1][vd * 32 + vj + 8] = vr1;
    }
    cur ^= 1;
  }

  // ---- epilogue: O /= l, store attn_vec at (t, b, h*64+d) row-major
#pragma unroll
  for (int t = 0; t < 4; ++t) {
#pragma unroll
    for (int r = 0; r < 8; ++r) {
      const int mloc = r + 8 * half;
      const int tok  = i0 + mloc;
      const int d    = t * 16 + lm;
      const float val = o[t][r] / lrun[r];
      AVws[((size_t)tok * BSZ + b) * DMODEL + hh * DHEAD + d] = f2bf(val);
    }
  }
}

// ---------------------------------------------------------------------------
// Kernel 3: out projection + residual.  resid[m,n] = h[m,n] + sum_k AV[m,k]*wo[n,k]
// Same double-buffered bf16 GEMM structure as kernel 1.
// ---------------------------------------------------------------------------
__global__ __launch_bounds__(256)
void out_proj_kernel(const unsigned short* __restrict__ AVws,
                     const unsigned short* __restrict__ wob,
                     const float* __restrict__ h,
                     float* __restrict__ resid)
{
  __shared__ unsigned short Abuf[2][128 * 32];
  __shared__ unsigned short Bbuf[2][64 * 32];
  const int tid  = threadIdx.x;
  const int lane = tid & 31, w = tid >> 5;
  const int lm   = lane & 15, half = lane >> 4;
  const int m0   = blockIdx.x * 128;
  const int n0   = blockIdx.y * 64;

  const int eA = tid * 16, mA = eA >> 5, kA = eA & 31;
  const int eB = tid * 8,  nB = eB >> 5, kB = eB & 31;
  const unsigned short* aSrc = AVws + (size_t)(m0 + mA) * DMODEL + kA;
  const unsigned short* bSrc = wob  + (size_t)(n0 + nB) * DMODEL + kB;

  f32x8 acc[4] = {};

  u32x4 ar0 = *(const u32x4*)(aSrc);
  u32x4 ar1 = *(const u32x4*)(aSrc + 8);
  u32x4 br0 = *(const u32x4*)(bSrc);
  *(u32x4*)&Abuf[0][mA * 32 + kA]     = ar0;
  *(u32x4*)&Abuf[0][mA * 32 + kA + 8] = ar1;
  *(u32x4*)&Bbuf[0][nB * 32 + kB]     = br0;

  int cur = 0;
  for (int k0 = 0; k0 < DMODEL; k0 += 32) {
    __syncthreads();
    const bool more = (k0 + 32) < DMODEL;
    if (more) {
      ar0 = *(const u32x4*)(aSrc + k0 + 32);
      ar1 = *(const u32x4*)(aSrc + k0 + 40);
      br0 = *(const u32x4*)(bSrc + k0 + 32);
    }

    FragBF a;
    const int mrow = w * 16 + lm;
    a.q[0] = *(const u32x4*)&Abuf[cur][mrow * 32 + half * 16];
    a.q[1] = *(const u32x4*)&Abuf[cur][mrow * 32 + half * 16 + 8];
#pragma unroll
    for (int t = 0; t < 4; ++t) {
      FragBF bf;
      const int n = t * 16 + lm;
      bf.q[0] = *(const u32x4*)&Bbuf[cur][n * 32 + half * 16];
      bf.q[1] = *(const u32x4*)&Bbuf[cur][n * 32 + half * 16 + 8];
      acc[t] = __builtin_amdgcn_wmma_f32_16x16x32_bf16(
          false, a.v, false, bf.v, (short)0, acc[t], false, false);
    }

    if (more) {
      *(u32x4*)&Abuf[cur ^ 1][mA * 32 + kA]     = ar0;
      *(u32x4*)&Abuf[cur ^ 1][mA * 32 + kA + 8] = ar1;
      *(u32x4*)&Bbuf[cur ^ 1][nB * 32 + kB]     = br0;
    }
    cur ^= 1;
  }

#pragma unroll
  for (int t = 0; t < 4; ++t) {
#pragma unroll
    for (int r = 0; r < 8; ++r) {
      const int mg = m0 + w * 16 + r + 8 * half;
      const int ng = n0 + t * 16 + lm;
      const size_t idx = (size_t)mg * DMODEL + ng;
      resid[idx] = h[idx] + acc[t][r];
    }
  }
}

// ---------------------------------------------------------------------------
// Kernel 4: LayerNorm over the last dim (1024), one block per row.
// ---------------------------------------------------------------------------
__global__ __launch_bounds__(256)
void layernorm_kernel(const float* __restrict__ resid,
                      const float* __restrict__ g,
                      const float* __restrict__ beta,
                      float* __restrict__ out)
{
  const int row = blockIdx.x;
  const int tid = threadIdx.x;
  const float* x = resid + (size_t)row * DMODEL;

  const f32x4 v = *(const f32x4*)(x + tid * 4);
  float s  = v.x + v.y + v.z + v.w;
  float s2 = v.x * v.x + v.y * v.y + v.z * v.z + v.w * v.w;
#pragma unroll
  for (int off = 16; off >= 1; off >>= 1) {
    s  += __shfl_xor(s,  off, 32);
    s2 += __shfl_xor(s2, off, 32);
  }
  __shared__ float red[2][8];
  const int wv = tid >> 5, ln = tid & 31;
  if (ln == 0) { red[0][wv] = s; red[1][wv] = s2; }
  __syncthreads();
  if (wv == 0) {
    float a  = red[0][ln & 7];
    float b2 = red[1][ln & 7];
#pragma unroll
    for (int off = 4; off >= 1; off >>= 1) {
      a  += __shfl_xor(a,  off, 32);
      b2 += __shfl_xor(b2, off, 32);
    }
    if (ln == 0) { red[0][0] = a; red[1][0] = b2; }
  }
  __syncthreads();
  const float mu  = red[0][0] * (1.0f / DMODEL);
  const float var = red[1][0] * (1.0f / DMODEL) - mu * mu;
  const float inv = rsqrtf(var + 1e-5f);

  const f32x4 gg = *(const f32x4*)(g + tid * 4);
  const f32x4 bb = *(const f32x4*)(beta + tid * 4);
  f32x4 o;
  o.x = (v.x - mu) * inv * gg.x + bb.x;
  o.y = (v.y - mu) * inv * gg.y + bb.y;
  o.z = (v.z - mu) * inv * gg.z + bb.z;
  o.w = (v.w - mu) * inv * gg.w + bb.w;
  *(f32x4*)(out + (size_t)row * DMODEL + tid * 4) = o;
}

// ---------------------------------------------------------------------------
// Launcher.  inputs: h, attn_mask, wq, wkv, wo, ln_g, ln_b
// Workspace: hb(16M) wqkv_b(6M) wo_b(2M) Q(16M) K(16M) V^T(16M) AV(16M)
//            resid(32M) = 120MB
// ---------------------------------------------------------------------------
extern "C" void kernel_launch(void* const* d_in, const int* in_sizes, int n_in,
                              void* d_out, int out_size, void* d_ws, size_t ws_size,
                              hipStream_t stream)
{
  const float*         h    = (const float*)d_in[0];
  const unsigned char* mask = (const unsigned char*)d_in[1];  // jnp.bool_: 1 byte/elem
  const float*         wq   = (const float*)d_in[2];
  const float*         wkv  = (const float*)d_in[3];
  const float*         wo   = (const float*)d_in[4];
  const float*         ln_g = (const float*)d_in[5];
  const float*         ln_b = (const float*)d_in[6];
  float*               out  = (float*)d_out;

  char* ws = (char*)d_ws;
  const size_t EH  = (size_t)NROWS * DMODEL;      // 8,388,608
  const size_t EWQ = (size_t)DMODEL * DMODEL;     // 1,048,576
  const size_t EWK = 2 * EWQ;                     // 2,097,152

  unsigned short* hb    = (unsigned short*)(ws);
  unsigned short* wqkvb = (unsigned short*)(ws + EH * 2);
  unsigned short* wob   = (unsigned short*)(ws + EH * 2 + (EWQ + EWK) * 2);
  char* p = ws + EH * 2 + (EWQ + EWK) * 2 + EWQ * 2;
  unsigned short* Qws  = (unsigned short*)(p);
  unsigned short* Kws  = (unsigned short*)(p + EH * 2);
  unsigned short* VTws = (unsigned short*)(p + EH * 4);
  unsigned short* AVws = (unsigned short*)(p + EH * 6);
  float*          res  = (float*)(p + EH * 8);

  // one-shot fp32 -> bf16 conversions (wq and wkv fused into one buffer)
  cvt_bf16_kernel<<<(int)(EH  / 2048), 256, 0, stream>>>(h,   hb,            (int)EH);
  cvt_bf16_kernel<<<(int)(EWQ / 2048), 256, 0, stream>>>(wq,  wqkvb,         (int)EWQ);
  cvt_bf16_kernel<<<(int)(EWK / 2048), 256, 0, stream>>>(wkv, wqkvb + EWQ,   (int)EWK);
  cvt_bf16_kernel<<<(int)(EWQ / 2048), 256, 0, stream>>>(wo,  wob,           (int)EWQ);

  qkv_proj_kernel<<<dim3(NROWS / 128, NQKV / 64), 256, 0, stream>>>(
      hb, wqkvb, Qws, Kws, VTws);
  flash_attn_kernel<<<dim3(T_TOK / 64, BSZ * NHEAD), 128, 0, stream>>>(
      Qws, Kws, VTws, mask, AVws);
  out_proj_kernel<<<dim3(NROWS / 128, DMODEL / 64), 256, 0, stream>>>(
      AVws, wob, h, res);
  layernorm_kernel<<<NROWS, 256, 0, stream>>>(res, ln_g, ln_b, out);
}
